// SelfAttention_31782757991072
// MI455X (gfx1250) — matrix operand order repaired and checked by reading, not compiled, not run on hardware
//
#include <hip/hip_runtime.h>

// ---------------------------------------------------------------------------
// Self-attention (B=8, C=256, H=W=64, inter=32), flash-attention style.
// bf16 WMMA (v_wmma_f32_16x16x32_bf16) with f32 accumulation, wave32.
// Round 4: vectorized float4 staging in the projection kernel (b64 LDS stores
// for the W tile); K/V attention tiles staged by the Tensor Data Mover
// (tensor_load_to_lds + s_wait_tensorcnt double-buffer pipeline).
// ---------------------------------------------------------------------------

typedef __bf16 bf16;
typedef __attribute__((ext_vector_type(16))) __bf16 v16bf;
typedef __attribute__((ext_vector_type(8)))  __bf16 v8bf;
typedef __attribute__((ext_vector_type(4)))  __bf16 v4bf;
typedef __attribute__((ext_vector_type(8)))  float  v8f;
typedef __attribute__((ext_vector_type(4)))  float  v4f;

#define B_  8
#define C_  256
#define N_  4096
#define D_  32
#define MT  64            // query tile (rows) per block
#define KT  64            // key tile per iteration
#define QK_STR 40         // Q/K LDS row stride in elements (80 B, 16B-aligned, conflict-free)
#define V_STR  72         // V LDS row stride in elements (144 B)
#define P_STR  72         // P LDS row stride in elements

#if defined(__has_builtin)
#if __has_builtin(__builtin_amdgcn_tensor_load_to_lds) && __has_builtin(__builtin_amdgcn_s_wait_tensorcnt)
#define HAVE_TDM 1
#endif
#endif

static __device__ __forceinline__ bf16 f2bf(float f) { return (bf16)f; }

static __device__ __forceinline__ v16bf cat16(const bf16* lo, const bf16* hi) {
  v8bf l = *(const v8bf*)lo;                  // ds_load_b128
  v8bf h = *(const v8bf*)hi;                  // ds_load_b128
  return __builtin_shufflevector(l, h, 0,1,2,3,4,5,6,7,8,9,10,11,12,13,14,15);
}

// A-matrix 16x32 bf16 per-lane layout (ISA 7.12.2): e<8 -> K=8*half+e ; e>=8 -> K=16+8*half+(e-8)
static __device__ __forceinline__ v16bf load_A(const bf16* rowbase, int half) {
  return cat16(rowbase + 8*half, rowbase + 16 + 8*half);
}
// B-matrix 32x16 bf16 per-lane layout (same as A): e<8 -> K=8*half+e ; e>=8 -> K=16+8*half+(e-8)
static __device__ __forceinline__ v16bf load_B(const bf16* colbase, int half) {
  return cat16(colbase + 8*half, colbase + 16 + 8*half);
}

static __device__ __forceinline__ v8f wmma_bf16(v16bf a, v16bf b, v8f c) {
  return __builtin_amdgcn_wmma_f32_16x16x32_bf16(false, a, false, b, (short)0, c, false, false);
}

#ifdef HAVE_TDM
typedef __attribute__((ext_vector_type(4))) unsigned int v4u;
typedef __attribute__((ext_vector_type(8))) int          v8i;
typedef __attribute__((ext_vector_type(4))) int          v4i;

static __device__ __forceinline__ unsigned lds_off(const void* p) {
  return (unsigned)(unsigned long long)p;     // low 32 bits of flat LDS addr = LDS offset
}

// 2-D tensor tile load, 2-byte elements, with LDS row padding (ISA 8.3-8.6).
// padi: pad_interval code (interval = 2^(padi+1) DWORDs); pada: pad_amount code (pada+1 DWORDs).
static __device__ __forceinline__ void tdm_load_2d(const void* gaddr, unsigned ldsa,
                                                   unsigned td0, unsigned td1,
                                                   unsigned tile0, unsigned tile1,
                                                   unsigned long long stride0,
                                                   unsigned padi, unsigned pada) {
  unsigned long long ga = (unsigned long long)gaddr;
  v4u g0;
  g0[0] = 1u;                                              // count=1, user descriptor
  g0[1] = ldsa;                                            // lds_addr (bytes)
  g0[2] = (unsigned)(ga & 0xFFFFFFFFu);                    // global_addr[31:0]
  g0[3] = (unsigned)((ga >> 32) & 0x1FFFFFFu) | (2u << 30);// global_addr[56:32], type=2
  v8i g1;
  g1[0] = (int)((1u << 16) | (1u << 20) | (padi << 22) | (pada << 25)); // data_size=2B, pad_enable
  g1[1] = (int)((td0 & 0xFFFFu) << 16);                    // tensor_dim0[15:0] @ bit48
  g1[2] = (int)(((td0 >> 16) & 0xFFFFu) | ((td1 & 0xFFFFu) << 16));
  g1[3] = (int)(((td1 >> 16) & 0xFFFFu) | (tile0 << 16));  // tile_dim0 @ bit112
  g1[4] = (int)tile1;                                      // tile_dim1 @ bit128, tile_dim2=0
  g1[5] = (int)(stride0 & 0xFFFFFFFFull);                  // tensor_dim0_stride[31:0] @ bit160
  g1[6] = (int)((stride0 >> 32) & 0xFFFFull);              // stride0[47:32]; stride1=0
  g1[7] = 0;
  v4i g2 = {0, 0, 0, 0};                                   // 2-D: groups 2/3 unused
  v4i g3 = {0, 0, 0, 0};
#if __clang_major__ >= 23
  v8i g4 = {0, 0, 0, 0, 0, 0, 0, 0};
  __builtin_amdgcn_tensor_load_to_lds(g0, g1, g2, g3, g4, 0);
#else
  __builtin_amdgcn_tensor_load_to_lds(g0, g1, g2, g3, 0);
#endif
}
#endif  // HAVE_TDM

// ---------------------------------------------------------------------------
// Kernel 1: fused projection GEMM.  Y[320 x N] = [Wq;Wk;Wv] @ X[b]  + bias.
// Rows 0..31 -> q_ws[b][n][d], 32..63 -> k_ws[b][n][d], 64..319 -> v_ws[b][c][n].
// ---------------------------------------------------------------------------
__global__ __launch_bounds__(256)
void proj_kernel(const float* __restrict__ x,
                 const float* __restrict__ Wq, const float* __restrict__ bq,
                 const float* __restrict__ Wk, const float* __restrict__ bk,
                 const float* __restrict__ Wv, const float* __restrict__ bv,
                 bf16* __restrict__ qw, bf16* __restrict__ kw, bf16* __restrict__ vw) {
  __shared__ __align__(16) bf16  Wl[320 * QK_STR];   // 25600 B
  __shared__ __align__(16) bf16  Xt[64 * QK_STR];    // 5120 B
  __shared__ float biasl[320];

  const int tid  = threadIdx.x;
  const int lane = tid & 31;
  const int wave = tid >> 5;
  const int idx  = lane & 15;
  const int half = lane >> 4;

  const int b  = blockIdx.x >> 6;
  const int nb = (blockIdx.x & 63) * MT;

  for (int i = tid; i < 320; i += 256)
    biasl[i] = (i < 32) ? bq[i] : (i < 64) ? bk[i - 32] : bv[i - 64];

  const int nct   = wave & 3;           // 16-wide n column tile
  const int rbase = (wave >> 2) * 10;   // 10 row-tiles per wave (20 total)

  v8f acc[10];
  #pragma unroll
  for (int i = 0; i < 10; ++i) acc[i] = v8f{0.f,0.f,0.f,0.f,0.f,0.f,0.f,0.f};

  for (int cc = 0; cc < C_; cc += 32) {
    __syncthreads();
    // stage W chunk (320 x 32) as bf16, float4 loads + b64 LDS stores
    #pragma unroll
    for (int i = 0; i < 10; ++i) {
      int id  = i * 256 + tid;          // 4-element group id (2560 groups)
      int row = id >> 3;                // 8 groups per 32-wide row
      int j4  = (id & 7) * 4;
      const float* src = (row < 32) ? (Wq + (size_t)row * C_)
                       : (row < 64) ? (Wk + (size_t)(row - 32) * C_)
                                    : (Wv + (size_t)(row - 64) * C_);
      v4f w4 = *(const v4f*)(src + cc + j4);
      v4bf w4b;
      #pragma unroll
      for (int e = 0; e < 4; ++e) w4b[e] = f2bf(w4[e]);
      *(v4bf*)(Wl + row * QK_STR + j4) = w4b;
    }
    // stage X transposed: Xt[n][ch] (64 x 32), float4 loads + b16 scatter
    #pragma unroll
    for (int i = 0; i < 2; ++i) {
      int id = i * 256 + tid;           // 512 groups of 4 consecutive n
      int n4 = (id & 15) * 4;
      int j  = id >> 4;                 // 0..31
      v4f x4 = *(const v4f*)(x + ((size_t)b * C_ + cc + j) * N_ + nb + n4);
      #pragma unroll
      for (int e = 0; e < 4; ++e) Xt[(n4 + e) * QK_STR + j] = f2bf(x4[e]);
    }
    __syncthreads();

    v16bf Bm = load_B(Xt + (nct * 16 + idx) * QK_STR, half);
    v16bf Am = load_A(Wl + (rbase * 16 + idx) * QK_STR, half);
    #pragma unroll
    for (int i = 0; i < 10; ++i) {
      v16bf An = Am;
      if (i < 9) An = load_A(Wl + ((rbase + i + 1) * 16 + idx) * QK_STR, half);
      acc[i] = wmma_bf16(Am, Bm, acc[i]);
      Am = An;
    }
  }
  __syncthreads();

  const int n = nb + nct * 16 + idx;
  #pragma unroll
  for (int i = 0; i < 10; ++i) {
    #pragma unroll
    for (int r = 0; r < 8; ++r) {
      int row = (rbase + i) * 16 + r + 8 * half;     // C-layout row
      bf16 vout = f2bf(acc[i][r] + biasl[row]);
      if (row < 32)
        qw[((size_t)b * N_ + n) * D_ + row] = vout;
      else if (row < 64)
        kw[((size_t)b * N_ + n) * D_ + (row - 32)] = vout;
      else
        vw[((size_t)b * C_ + (row - 64)) * N_ + n] = vout;
    }
  }
}

// ---------------------------------------------------------------------------
// Kernel 2: flash attention.  Grid: 512 blocks (b, 64-query tile), 256 threads.
// Wave w: row-tile rt=w&3 (16 queries), c-columns cgrp=(w>>2)*128 .. +128.
// Double-buffered K/V tiles (96 KB LDS total, CDNA5 WGP has 320 KB).
// ---------------------------------------------------------------------------
__global__ __launch_bounds__(256)
void attn_kernel(const bf16* __restrict__ qw, const bf16* __restrict__ kw,
                 const bf16* __restrict__ vw,
                 const float* __restrict__ x, const float* __restrict__ gammap,
                 float* __restrict__ out) {
  __shared__ __align__(16) bf16 Qs[MT * QK_STR];          //  5120 B
  __shared__ __align__(16) bf16 Ks[2][KT * QK_STR];       // 10240 B
  __shared__ __align__(16) bf16 Vs[2][C_ * V_STR];        // 73728 B
  __shared__ __align__(16) bf16 Ps[4 * 16 * P_STR];       //  9216 B (one buffer per row-tile)

  const int tid  = threadIdx.x;
  const int lane = tid & 31;
  const int wave = tid >> 5;
  const int idx  = lane & 15;
  const int half = lane >> 4;
  const int rt   = wave & 3;
  const int cgrp = wave >> 2;

  const int b  = blockIdx.x >> 6;
  const int qb = (blockIdx.x & 63) * MT;

  // per-thread staging coordinates
  const int sr  = tid >> 2, sch = tid & 3;     // K/Q tile: row, 8-elem chunk

  // stage Q tile (64 x 32), one b128 per thread
  *(v8bf*)(Qs + sr * QK_STR + sch * 8) =
      *(const v8bf*)(qw + ((size_t)b * N_ + qb + sr) * D_ + sch * 8);
  __syncthreads();

  const v16bf qa = load_A(Qs + (rt * 16 + idx) * QK_STR, half);
  const float gamma = gammap[0];

  v8f acc[8];
  #pragma unroll
  for (int i = 0; i < 8; ++i) acc[i] = v8f{0.f,0.f,0.f,0.f,0.f,0.f,0.f,0.f};
  float mrow[8], lrow[8];
  #pragma unroll
  for (int r = 0; r < 8; ++r) { mrow[r] = -__builtin_inff(); lrow[r] = 0.f; }

  const int nIter = N_ / KT;                 // 64

#ifdef HAVE_TDM
  // ---- Tensor Data Mover staging: wave 0 issues one DMA per tile ----
  // K tile: 64 rows x 32 bf16 (64B rows), pad 16B/row  -> stride QK_STR
  // V tile: 256 rows x 64 bf16 (128B rows), pad 16B/row -> stride V_STR
  auto issue_tile = [&](int kb, int buf) {
    if (wave == 0) {
      tdm_load_2d(kw + ((size_t)b * N_ + kb) * D_, lds_off(Ks[buf]),
                  D_, KT, D_, KT, (unsigned long long)D_, /*padi=*/3, /*pada=*/3);
      tdm_load_2d(vw + (size_t)b * C_ * N_ + kb, lds_off(Vs[buf]),
                  KT, C_, KT, C_, (unsigned long long)N_, /*padi=*/4, /*pada=*/3);
    }
  };
  issue_tile(0, 0);
#else
  // ---- fallback: register prefetch of one K/V tile ----
  v8bf kreg;
  v8bf vreg[8];
  auto prefetch = [&](int kb) {
    kreg = *(const v8bf*)(kw + ((size_t)b * N_ + kb + sr) * D_ + sch * 8);
    #pragma unroll
    for (int i = 0; i < 8; ++i) {
      int id2 = i * 256 + tid;
      int c = id2 >> 3, ch = id2 & 7;
      vreg[i] = *(const v8bf*)(vw + ((size_t)b * C_ + c) * N_ + kb + ch * 8);
    }
  };
  auto store_tile = [&](int buf) {
    *(v8bf*)(Ks[buf] + sr * QK_STR + sch * 8) = kreg;
    #pragma unroll
    for (int i = 0; i < 8; ++i) {
      int id2 = i * 256 + tid;
      int c = id2 >> 3, ch = id2 & 7;
      *(v8bf*)(Vs[buf] + c * V_STR + ch * 8) = vreg[i];
    }
  };
  prefetch(0);
  store_tile(0);
#endif

  for (int it = 0; it < nIter; ++it) {
    const int cur = it & 1;

#ifdef HAVE_TDM
    if (wave == 0) __builtin_amdgcn_s_wait_tensorcnt(0);  // tile `cur` landed in LDS
    __syncthreads();   // publish buf `cur`; all waves done with buf cur^1 and prev P
    if (it + 1 < nIter) issue_tile((it + 1) * KT, cur ^ 1);  // DMA overlaps compute
#else
    if (it + 1 < nIter) prefetch((it + 1) * KT);   // global loads in flight
    __syncthreads();   // buf `cur` staged by all waves; prev P reads done
#endif

    // S strip: 16 queries x 64 keys = 4 WMMAs (K = d = 32), pipelined B loads
    v8f st[4];
    {
      const bf16* kb0 = Ks[cur] + idx * QK_STR;
      v16bf kB = load_B(kb0, half);
      #pragma unroll
      for (int kt = 0; kt < 4; ++kt) {
        v16bf kN = kB;
        if (kt < 3) kN = load_B(kb0 + (kt + 1) * 16 * QK_STR, half);
        st[kt] = wmma_bf16(qa, kB, v8f{0.f,0.f,0.f,0.f,0.f,0.f,0.f,0.f});
        kB = kN;
      }
    }

    // online softmax (per C-layout row r -> query rt*16 + r + 8*half)
    float alpha[8];
    #pragma unroll
    for (int r = 0; r < 8; ++r) {
      float tm = fmaxf(fmaxf(st[0][r], st[1][r]), fmaxf(st[2][r], st[3][r]));
      tm = fmaxf(tm, __shfl_xor(tm, 1, 32));
      tm = fmaxf(tm, __shfl_xor(tm, 2, 32));
      tm = fmaxf(tm, __shfl_xor(tm, 4, 32));
      tm = fmaxf(tm, __shfl_xor(tm, 8, 32));
      float mnew = fmaxf(mrow[r], tm);
      float a = __expf(mrow[r] - mnew);
      mrow[r] = mnew;
      float ps = 0.f;
      #pragma unroll
      for (int kt = 0; kt < 4; ++kt) {
        float p = __expf(st[kt][r] - mnew);
        st[kt][r] = p;
        ps += p;
      }
      ps += __shfl_xor(ps, 1, 32);
      ps += __shfl_xor(ps, 2, 32);
      ps += __shfl_xor(ps, 4, 32);
      ps += __shfl_xor(ps, 8, 32);
      lrow[r] = lrow[r] * a + ps;
      alpha[r] = a;
    }
    #pragma unroll
    for (int ct = 0; ct < 8; ++ct)
      #pragma unroll
      for (int r = 0; r < 8; ++r) acc[ct][r] *= alpha[r];

    // P (bf16) to LDS; waves w and w+4 compute identical P -> only cgrp==0 writes
    if (cgrp == 0) {
      bf16* pw = Ps + rt * 16 * P_STR;
      #pragma unroll
      for (int kt = 0; kt < 4; ++kt)
        #pragma unroll
        for (int r = 0; r < 8; ++r)
          pw[(r + 8 * half) * P_STR + kt * 16 + idx] = f2bf(st[kt][r]);
    }
    __syncthreads();   // publish P

    // read P in A layout (2 K-chunks of 32 keys)
    const bf16* pb = Ps + (rt * 16 + idx) * P_STR;   // row = query-local = idx
    v16bf pa0 = load_A(pb, half);
    v16bf pa1 = load_A(pb + 32, half);

    // O += P @ V : 8 c-tiles x 2 K-chunks, pipelined B loads
    {
      const bf16* vbase = Vs[cur] + (cgrp * 128 + idx) * V_STR;
      v16bf b0 = load_B(vbase, half);
      v16bf b1 = load_B(vbase + 32, half);
      #pragma unroll
      for (int ct = 0; ct < 8; ++ct) {
        v16bf n0 = b0, n1 = b1;
        if (ct < 7) {
          const bf16* vb = vbase + (ct + 1) * 16 * V_STR;
          n0 = load_B(vb, half);
          n1 = load_B(vb + 32, half);
        }
        acc[ct] = wmma_bf16(pa0, b0, acc[ct]);
        acc[ct] = wmma_bf16(pa1, b1, acc[ct]);
        b0 = n0; b1 = n1;
      }
    }

#ifndef HAVE_TDM
    // stage next tile into the other buffer (waits on prefetch loadcnt here)
    if (it + 1 < nIter) store_tile(cur ^ 1);
#endif
  }

  // epilogue: out = gamma * O / l + x  (rows of C-layout tile are contiguous in m)
  float inv[8];
  #pragma unroll
  for (int r = 0; r < 8; ++r) inv[r] = gamma / lrow[r];
  const int m0 = qb + rt * 16 + 8 * half;
  #pragma unroll
  for (int ct = 0; ct < 8; ++ct) {
    int c = cgrp * 128 + ct * 16 + idx;
    size_t off = ((size_t)b * C_ + c) * (size_t)N_ + (size_t)m0;
    v4f x0 = *(const v4f*)(x + off);
    v4f x1 = *(const v4f*)(x + off + 4);
    v4f o0, o1;
    #pragma unroll
    for (int r = 0; r < 4; ++r) o0[r] = acc[ct][r] * inv[r] + x0[r];
    #pragma unroll
    for (int r = 0; r < 4; ++r) o1[r] = acc[ct][r + 4] * inv[r + 4] + x1[r];
    *(v4f*)(out + off) = o0;
    *(v4f*)(out + off + 4) = o1;
  }
}

// ---------------------------------------------------------------------------
extern "C" void kernel_launch(void* const* d_in, const int* in_sizes, int n_in,
                              void* d_out, int out_size, void* d_ws, size_t ws_size,
                              hipStream_t stream) {
  (void)in_sizes; (void)n_in; (void)out_size; (void)ws_size;
  const float* x     = (const float*)d_in[0];
  const float* Wq    = (const float*)d_in[1];
  const float* bq    = (const float*)d_in[2];
  const float* Wk    = (const float*)d_in[3];
  const float* bk    = (const float*)d_in[4];
  const float* Wv    = (const float*)d_in[5];
  const float* bv    = (const float*)d_in[6];
  const float* gamma = (const float*)d_in[7];
  float* out = (float*)d_out;

  // workspace layout (bf16): q [8][4096][32], k [8][4096][32], v [8][256][4096]
  bf16* qw = (bf16*)d_ws;
  bf16* kw = qw + (size_t)B_ * N_ * D_;
  bf16* vw = kw + (size_t)B_ * N_ * D_;   // spans 16 MB, total 20 MB

  dim3 grid(B_ * (N_ / MT));   // 512
  dim3 block(256);
  proj_kernel<<<grid, block, 0, stream>>>(x, Wq, bq, Wk, bk, Wv, bv, qw, kw, vw);
  attn_kernel<<<grid, block, 0, stream>>>(qw, kw, vw, x, gamma, out);
}
